// Edge_aware_unit_2877628088638
// MI455X (gfx1250) — matrix operand order repaired and checked by reading, not compiled, not run on hardware
//
#include <hip/hip_runtime.h>

// MI455X / gfx1250, wave32.
// conv1x1: V_WMMA_F32_16X16X4_F32 (fp32 matrix op, small layer).
// conv3x3 layers: bf16x3 split-precision on V_WMMA_F32_16X16X32_BF16
//   (x = hi + lo in bf16; A*B ~= Ah*Bh + Ah*Bl + Al*Bh, fp32 accumulate)
//   -> ~2.7x the fp32 matrix rate at fp32-class accuracy.

typedef __attribute__((ext_vector_type(2)))  float        v2f;
typedef __attribute__((ext_vector_type(8)))  float        v8f;
typedef __attribute__((ext_vector_type(16))) __bf16       v16bf;
typedef __attribute__((ext_vector_type(4)))  unsigned int u4;

#define EPS 1e-5f

#define BN_ 16
#define H_  128
#define W_  128
#define HW_ (H_ * W_)
#define CIN_ 64
#define C1_  256
#define C2_  128
#define C3_  64

// ---- workspace layout ----
// floats: [0,1024) folded BN params
// bf16 (starting at float offset 1024): w2t_hi[294912], w2t_lo[294912],
//                                       w3t_hi[73728],  w3t_lo[73728]
// floats: c1 at 369664 (67108864), c2 after c1 (33554432); edge reuses c1.
#define P_SCALE1 0
#define P_BIAS1  256
#define P_SCALE2 512
#define P_BIAS2  640
#define P_SCALE3 768
#define P_BIAS3  832
#define W2T_ELEMS (9 * 128 * 256)
#define W3T_ELEMS (9 * 64 * 128)
#define C1_OFF   (1024 + W2T_ELEMS + W3T_ELEMS)          // 369664 floats
#define C1_ELEMS ((size_t)BN_ * C1_ * HW_)               // 67108864
#define C2_OFF   ((size_t)C1_OFF + C1_ELEMS)
#define C2_ELEMS ((size_t)BN_ * C2_ * HW_)
// total ws: (C2_OFF + C2_ELEMS)*4 ~= 404 MB

// ---------------- helpers ----------------------------------------------------
__device__ inline v16bf load_a_frag(const __bf16* p) {
  // A 16x32 bf16 fragment: per-lane K = base+0..7 (v0-3) and base+16..23 (v4-7)
  v16bf a;
  ((u4*)&a)[0] = *(const u4*)(p);
  ((u4*)&a)[1] = *(const u4*)(p + 16);
  return a;
}
__device__ inline v16bf load_b_frag(const __bf16* p) {
  // B 32x16 bf16 fragment: per-lane K = base+0..7 (v0-3) and base+16..23 (v4-7)
  v16bf b;
  ((u4*)&b)[0] = *(const u4*)(p);
  ((u4*)&b)[1] = *(const u4*)(p + 16);
  return b;
}

// ---------------- BN fold: scale = g*rsqrt(v+eps); bias = (b-m)*scale + be ----
__global__ void fold_params_kernel(
    const float* __restrict__ b1, const float* __restrict__ g1, const float* __restrict__ be1,
    const float* __restrict__ m1, const float* __restrict__ v1,
    const float* __restrict__ b2, const float* __restrict__ g2, const float* __restrict__ be2,
    const float* __restrict__ m2, const float* __restrict__ v2,
    const float* __restrict__ b3, const float* __restrict__ g3, const float* __restrict__ be3,
    const float* __restrict__ m3, const float* __restrict__ v3,
    float* __restrict__ ws) {
  int i = threadIdx.x;  // 256 threads
  if (i < 256) {
    float s = g1[i] * rsqrtf(v1[i] + EPS);
    ws[P_SCALE1 + i] = s;
    ws[P_BIAS1 + i] = (b1[i] - m1[i]) * s + be1[i];
  }
  if (i < 128) {
    float s = g2[i] * rsqrtf(v2[i] + EPS);
    ws[P_SCALE2 + i] = s;
    ws[P_BIAS2 + i] = (b2[i] - m2[i]) * s + be2[i];
  }
  if (i < 64) {
    float s = g3[i] * rsqrtf(v3[i] + EPS);
    ws[P_SCALE3 + i] = s;
    ws[P_BIAS3 + i] = (b3[i] - m3[i]) * s + be3[i];
  }
}

// ------- weight reorder + bf16 hi/lo split: [co][ci][3][3] -> [tap][co][ci] ---
__global__ void reorder_w_split_kernel(const float* __restrict__ w,
                                       __bf16* __restrict__ whi,
                                       __bf16* __restrict__ wlo,
                                       int co_n, int ci_n) {
  int i = blockIdx.x * 256 + threadIdx.x;
  int total = co_n * ci_n * 9;
  if (i >= total) return;
  int tap = i % 9;
  int t = i / 9;
  int ci = t % ci_n;
  int co = t / ci_n;
  float v = w[i];
  __bf16 h = (__bf16)v;
  size_t o = ((size_t)tap * co_n + co) * ci_n + ci;
  whi[o] = h;
  wlo[o] = (__bf16)(v - (float)h);
}

// ---------------- concat: copy input into out[:, 0:64] ------------------------
__global__ void copy_input_kernel(const float4* __restrict__ in, float* __restrict__ out) {
  int i = blockIdx.x * 256 + threadIdx.x;  // total = 16*64*16384/4
  if (i >= BN_ * CIN_ * HW_ / 4) return;
  int p4 = i & (HW_ / 4 - 1);
  int plane = i >> 12;  // b*64 + c
  int b = plane >> 6;
  int c = plane & 63;
  ((float4*)out)[((size_t)(b * 128 + c)) * (HW_ / 4) + p4] = in[i];
}

// ---------------- conv1x1 64->256 via WMMA f32 16x16x4 ------------------------
__global__ __launch_bounds__(256) void conv1x1_wmma_kernel(
    const float* __restrict__ x, const float* __restrict__ w1,
    const float* __restrict__ ws, float* __restrict__ c1) {
  __shared__ float xs[16 * 68];  // [n=16][ci=64 pad 68]
  int tid = threadIdx.x;
  int hw0 = blockIdx.x * 16;
  int b = blockIdx.y;
  {
    int n = tid & 15, cig = tid >> 4;
    const float* xb = x + (size_t)b * CIN_ * HW_ + hw0 + n;
#pragma unroll
    for (int r = 0; r < 4; ++r) {
      int ci = cig * 4 + r;
      xs[n * 68 + ci] = xb[(size_t)ci * HW_];
    }
  }
  __syncthreads();

  int lane = tid & 31, half = lane >> 4, lm = lane & 15;
  int wave = tid >> 5;
  int co_base = blockIdx.z * 128 + wave * 16;

  const float* arow = w1 + (size_t)(co_base + lm) * CIN_ + 2 * half;
  const float* brow = &xs[lm * 68 + 2 * half];

  v8f acc = {};
#pragma unroll
  for (int k0 = 0; k0 < CIN_; k0 += 4) {
    v2f a = *(const v2f*)(arow + k0);
    v2f bb = *(const v2f*)(brow + k0);
    acc = __builtin_amdgcn_wmma_f32_16x16x4_f32(false, a, false, bb, (short)0, acc,
                                                false, false);
  }
#pragma unroll
  for (int v = 0; v < 8; ++v) {
    int co = co_base + v + half * 8;
    float y = fmaxf(acc[v] * ws[P_SCALE1 + co] + ws[P_BIAS1 + co], 0.f);
    c1[((size_t)(b * C1_ + co)) * HW_ + hw0 + lm] = y;
  }
}

// -------- generic 3x3 conv via bf16x3 split WMMA (CIN in {256,128}) ----------
// block = CIN threads (ci = tid for LDS fill), waves = CIN/32, each wave 16 co,
// COUT = CIN/2. grid: (H * 8 pixel-segments, B). Activation hi/lo split done
// on the fly while staging into two bf16 LDS tiles (channel-innermost, +8 pad
// keeps every 32B fragment read 16B-aligned and bank-spread).
template <int CIN, int CIP>
__global__ __launch_bounds__(CIN) void conv3x3_bf16x3_kernel(
    const float* __restrict__ xin,
    const __bf16* __restrict__ wthi, const __bf16* __restrict__ wtlo,
    const float* __restrict__ scale, const float* __restrict__ bias,
    float* __restrict__ out, int out_nch, int out_choff) {
  constexpr int COUT = CIN / 2;
  __shared__ __align__(16) __bf16 ldsh[3 * 18 * CIP];
  __shared__ __align__(16) __bf16 ldsl[3 * 18 * CIP];
  int tid = threadIdx.x;
  int h = blockIdx.x >> 3;
  int w0 = (blockIdx.x & 7) * 16;
  int b = blockIdx.y;

  // stage 3x18xCIN halo tile, splitting fp32 -> bf16 hi/lo
  {
    const float* src = xin + ((size_t)(b * CIN + tid)) * HW_;
#pragma unroll
    for (int r = 0; r < 3; ++r) {
      int hr = h + r - 1;
      bool rowok = (unsigned)hr < (unsigned)H_;
      const float* rp = src + hr * W_;
#pragma unroll
      for (int c = 0; c < 18; ++c) {
        int wc = w0 - 1 + c;
        float v = (rowok && (unsigned)wc < (unsigned)W_) ? rp[wc] : 0.f;
        __bf16 hi = (__bf16)v;
        ldsh[(r * 18 + c) * CIP + tid] = hi;
        ldsl[(r * 18 + c) * CIP + tid] = (__bf16)(v - (float)hi);
      }
    }
  }
  __syncthreads();

  int lane = tid & 31, half = lane >> 4, lm = lane & 15;
  int wave = tid >> 5;
  int co_base = wave * 16;

  v8f acc = {};
#pragma unroll
  for (int tap = 0; tap < 9; ++tap) {
    int kh = tap / 3, kw = tap % 3;
    size_t wrow = ((size_t)(tap * COUT + co_base + lm)) * CIN + half * 8;
    const __bf16* ah_row = wthi + wrow;
    const __bf16* al_row = wtlo + wrow;
    int bidx = (kh * 18 + kw + lm) * CIP + half * 8;
    const __bf16* bh_col = &ldsh[bidx];
    const __bf16* bl_col = &ldsl[bidx];
#pragma unroll 4
    for (int k0 = 0; k0 < CIN; k0 += 32) {
      v16bf ah = load_a_frag(ah_row + k0);
      v16bf al = load_a_frag(al_row + k0);
      v16bf bh = load_b_frag(bh_col + k0);
      v16bf bl = load_b_frag(bl_col + k0);
      acc = __builtin_amdgcn_wmma_f32_16x16x32_bf16(false, ah, false, bh, (short)0,
                                                    acc, false, false);
      acc = __builtin_amdgcn_wmma_f32_16x16x32_bf16(false, ah, false, bl, (short)0,
                                                    acc, false, false);
      acc = __builtin_amdgcn_wmma_f32_16x16x32_bf16(false, al, false, bh, (short)0,
                                                    acc, false, false);
    }
  }
#pragma unroll
  for (int v = 0; v < 8; ++v) {
    int co = co_base + v + half * 8;
    float y = fmaxf(acc[v] * scale[co] + bias[co], 0.f);
    out[((size_t)(b * out_nch + out_choff + co)) * HW_ + h * W_ + w0 + lm] = y;
  }
}

// ---------------- depthwise sobel (summed kernel), pad=1 ----------------------
__global__ void sobel_kernel(const float* __restrict__ c2, float* __restrict__ edge) {
  int i = blockIdx.x * 256 + threadIdx.x;  // total = 16*128*16384
  if (i >= BN_ * C2_ * HW_) return;
  int w = i & (W_ - 1);
  int h = (i >> 7) & (H_ - 1);
  const float* p = c2 + (((size_t)i >> 14) << 14);
  const float k[3][3] = {{2.f, 4.f, 4.f}, {-2.f, 0.f, 2.f}, {-4.f, -4.f, -2.f}};
  float acc = 0.f;
#pragma unroll
  for (int r = 0; r < 3; ++r) {
    int hr = h + r - 1;
    if ((unsigned)hr >= (unsigned)H_) continue;
#pragma unroll
    for (int c = 0; c < 3; ++c) {
      int wc = w + c - 1;
      if ((unsigned)wc >= (unsigned)W_) continue;
      acc += k[r][c] * p[hr * W_ + wc];
    }
  }
  edge[i] = acc;
}

extern "C" void kernel_launch(void* const* d_in, const int* in_sizes, int n_in,
                              void* d_out, int out_size, void* d_ws, size_t ws_size,
                              hipStream_t stream) {
  const float* input_f = (const float*)d_in[0];
  const float* w1 = (const float*)d_in[1];
  const float* b1 = (const float*)d_in[2];
  const float* g1 = (const float*)d_in[3];
  const float* be1 = (const float*)d_in[4];
  const float* m1 = (const float*)d_in[5];
  const float* v1 = (const float*)d_in[6];
  const float* w2 = (const float*)d_in[7];
  const float* b2 = (const float*)d_in[8];
  const float* g2 = (const float*)d_in[9];
  const float* be2 = (const float*)d_in[10];
  const float* m2 = (const float*)d_in[11];
  const float* v2 = (const float*)d_in[12];
  const float* w3 = (const float*)d_in[13];
  const float* b3 = (const float*)d_in[14];
  const float* g3 = (const float*)d_in[15];
  const float* be3 = (const float*)d_in[16];
  const float* m3 = (const float*)d_in[17];
  const float* v3 = (const float*)d_in[18];

  float* ws = (float*)d_ws;
  float* out = (float*)d_out;
  __bf16* w2t_hi = (__bf16*)(ws + 1024);
  __bf16* w2t_lo = w2t_hi + W2T_ELEMS;
  __bf16* w3t_hi = w2t_lo + W2T_ELEMS;
  __bf16* w3t_lo = w3t_hi + W3T_ELEMS;
  float* c1 = ws + C1_OFF;
  float* c2 = ws + C2_OFF;
  float* edge = ws + C1_OFF;  // reuses c1 region (dead after conv2)

  // 1) fold BN params
  fold_params_kernel<<<1, 256, 0, stream>>>(b1, g1, be1, m1, v1, b2, g2, be2, m2, v2,
                                            b3, g3, be3, m3, v3, ws);
  // 2) transpose + bf16 hi/lo split 3x3 weights to [tap][co][ci]
  reorder_w_split_kernel<<<(W2T_ELEMS + 255) / 256, 256, 0, stream>>>(w2, w2t_hi,
                                                                      w2t_lo, C2_, C1_);
  reorder_w_split_kernel<<<(W3T_ELEMS + 255) / 256, 256, 0, stream>>>(w3, w3t_hi,
                                                                      w3t_lo, C3_, C2_);
  // 3) concat part 1: input -> out[:, 0:64]
  copy_input_kernel<<<(BN_ * CIN_ * HW_ / 4 + 255) / 256, 256, 0, stream>>>(
      (const float4*)input_f, out);
  // 4) conv1x1 + BN + ReLU -> c1 (fp32 WMMA)
  conv1x1_wmma_kernel<<<dim3(HW_ / 16, BN_, 2), 256, 0, stream>>>(input_f, w1, ws, c1);
  // 5) conv3x3 256->128 + BN + ReLU -> c2 (bf16x3 WMMA)
  conv3x3_bf16x3_kernel<256, 264><<<dim3(H_ * 8, BN_), 256, 0, stream>>>(
      c1, w2t_hi, w2t_lo, ws + P_SCALE2, ws + P_BIAS2, c2, C2_, 0);
  // 6) sobel depthwise -> edge (overwrites c1 region)
  sobel_kernel<<<(BN_ * C2_ * HW_ + 255) / 256, 256, 0, stream>>>(c2, edge);
  // 7) conv3x3 128->64 + BN + ReLU -> out[:, 64:128] (bf16x3 WMMA)
  conv3x3_bf16x3_kernel<128, 136><<<dim3(H_ * 8, BN_), 128, 0, stream>>>(
      edge, w3t_hi, w3t_lo, ws + P_SCALE3, ws + P_BIAS3, out, 128, 64);
}
